// GINEncoder_36764920054230
// MI455X (gfx1250) — matrix operand order repaired and checked by reading, not compiled, not run on hardware
//
#include <hip/hip_runtime.h>
#include <hip/hip_bf16.h>
#include <stddef.h>

typedef __attribute__((ext_vector_type(16))) _Float16 v16h;
typedef __attribute__((ext_vector_type(8)))  _Float16 v8h;
typedef __attribute__((ext_vector_type(8)))  float    v8f;

#define NNODES 50000
#define NEDGES 640000
#define HID    128
#define NL     3
#define NG     64
#define NC     2

__device__ __forceinline__ void atomAddF(float* p, float v) {
    __hip_atomic_fetch_add(p, v, __ATOMIC_RELAXED, __HIP_MEMORY_SCOPE_AGENT);
}

// out[i] = in[i], vectorized by float4
__global__ __launch_bounds__(256) void copy_vec4(const float* __restrict__ in,
                                                 float* __restrict__ out, int n4) {
    int i = blockIdx.x * 256 + threadIdx.x;
    if (i < n4) ((float4*)out)[i] = ((const float4*)in)[i];
}

// One wave per edge: agg[dst] += x[src] (128 floats, 4 per lane).
__global__ __launch_bounds__(256) void scatter_edges(const float* __restrict__ x,
                                                     const int* __restrict__ src,
                                                     const int* __restrict__ dst,
                                                     float* __restrict__ agg, int nEdges) {
    int tid  = blockIdx.x * 256 + threadIdx.x;
    int e    = tid >> 5;
    int lane = tid & 31;
    if (e >= nEdges) return;
    int s = src[e], d = dst[e];                    // uniform per wave -> scalar loads
    float4 v = *(const float4*)(x + (size_t)s * HID + lane * 4);
    float* p = agg + (size_t)d * HID + lane * 4;
    atomAddF(p + 0, v.x); atomAddF(p + 1, v.y);
    atomAddF(p + 2, v.z); atomAddF(p + 3, v.w);
}

// Fused GIN MLP layer: out = relu( relu(in @ W1 + b1) @ W2 + b2 )
// Block: 256 threads = 8 waves; 64 rows per block; wave w owns N-tile [16w,16w+16).
__global__ __launch_bounds__(256) void gin_mlp(const float* __restrict__ in,
                                               float* __restrict__ out,
                                               const float* __restrict__ W1,
                                               const float* __restrict__ B1,
                                               const float* __restrict__ W2,
                                               const float* __restrict__ B2,
                                               int nRows) {
    __shared__ _Float16 sW1[HID * HID];   // transposed: sW1[n*128+k] = W1[k][n]
    __shared__ _Float16 sW2[HID * HID];
    __shared__ _Float16 sIn[64 * HID];    // row-major activation tile
    __shared__ _Float16 sH [64 * HID];    // hidden layer tile
    __shared__ float    sB1[HID], sB2[HID];

    const int t       = threadIdx.x;
    const int lane    = t & 31;
    const int wv      = t >> 5;
    const int rowBase = blockIdx.x * 64;

    for (int idx = t; idx < HID * HID; idx += 256) {
        int k = idx >> 7, n = idx & 127;
        sW1[n * HID + k] = (_Float16)W1[idx];
        sW2[n * HID + k] = (_Float16)W2[idx];
    }
    if (t < HID) { sB1[t] = B1[t]; sB2[t] = B2[t]; }
    for (int idx = t; idx < 64 * HID; idx += 256) {
        int r = idx >> 7, c = idx & 127;
        int row = rowBase + r;
        sIn[idx] = (_Float16)((row < nRows) ? in[(size_t)row * HID + c] : 0.0f);
    }
    __syncthreads();

    const int cl      = lane & 15;
    const int hiHalf  = lane >> 4;            // 0: lanes 0-15, 1: lanes 16-31
    const int col     = wv * 16 + cl;         // output column owned by this lane
    const float bias1 = sB1[col];
    const float bias2 = sB2[col];
    const int aOff    = hiHalf ? 16 : 0;      // A-fragment K offset
    const int bOff    = hiHalf ? 16 : 0;      // B-fragment K offset
    const int rOffL   = hiHalf * 8;           // C/D row offset for this lane group

    // ---- GEMM1: sH = relu(sIn @ W1 + b1) ----
    for (int m = 0; m < 4; ++m) {
        v8f acc = {};
        const _Float16* aRow = &sIn[(m * 16 + cl) * HID + aOff];
        const _Float16* bCol = &sW1[col * HID + bOff];
        #pragma unroll
        for (int ks = 0; ks < 4; ++ks) {
            v8h lo = *(const v8h*)(aRow + ks * 32);
            v8h hi = *(const v8h*)(aRow + ks * 32 + 8);
            v16h a;
            #pragma unroll
            for (int i = 0; i < 8; ++i) { a[i] = lo[i]; a[8 + i] = hi[i]; }
            v16h b = *(const v16h*)(bCol + ks * 32);
            acc = __builtin_amdgcn_wmma_f32_16x16x32_f16(false, a, false, b,
                                                         (short)0, acc, false, false);
        }
        #pragma unroll
        for (int i = 0; i < 8; ++i) {
            float h = acc[i] + bias1;
            sH[(m * 16 + rOffL + i) * HID + col] = (_Float16)(h > 0.f ? h : 0.f);
        }
    }
    __syncthreads();

    // ---- GEMM2: out = relu(sH @ W2 + b2) ----
    for (int m = 0; m < 4; ++m) {
        v8f acc = {};
        const _Float16* aRow = &sH[(m * 16 + cl) * HID + aOff];
        const _Float16* bCol = &sW2[col * HID + bOff];
        #pragma unroll
        for (int ks = 0; ks < 4; ++ks) {
            v8h lo = *(const v8h*)(aRow + ks * 32);
            v8h hi = *(const v8h*)(aRow + ks * 32 + 8);
            v16h a;
            #pragma unroll
            for (int i = 0; i < 8; ++i) { a[i] = lo[i]; a[8 + i] = hi[i]; }
            v16h b = *(const v16h*)(bCol + ks * 32);
            acc = __builtin_amdgcn_wmma_f32_16x16x32_f16(false, a, false, b,
                                                         (short)0, acc, false, false);
        }
        #pragma unroll
        for (int i = 0; i < 8; ++i) {
            int row = rowBase + m * 16 + rOffL + i;
            if (row < nRows) {
                float o = acc[i] + bias2;
                out[(size_t)row * HID + col] = o > 0.f ? o : 0.f;
            }
        }
    }
}

__global__ __launch_bounds__(256) void pool_zero(float* __restrict__ sums,
                                                 float* __restrict__ counts) {
    int i = blockIdx.x * 256 + threadIdx.x;
    if (i < NG * HID) sums[i] = 0.f;
    if (i < NG) counts[i] = 0.f;
}

// One wave per node: sums[batch[node]] += x[node]; counts[batch[node]] += 1
__global__ __launch_bounds__(256) void pool_scatter(const float* __restrict__ x,
                                                    const int* __restrict__ batch,
                                                    float* __restrict__ sums,
                                                    float* __restrict__ counts, int nNodes) {
    int tid  = blockIdx.x * 256 + threadIdx.x;
    int node = tid >> 5;
    int lane = tid & 31;
    if (node >= nNodes) return;
    int g = batch[node];
    float4 v = *(const float4*)(x + (size_t)node * HID + lane * 4);
    float* p = sums + (size_t)g * HID + lane * 4;
    atomAddF(p + 0, v.x); atomAddF(p + 1, v.y);
    atomAddF(p + 2, v.z); atomAddF(p + 3, v.w);
    if (lane == 0) atomAddF(&counts[g], 1.0f);
}

// Single block: g = sums/max(counts,1); logits = g @ head_w + head_b
// d_out layout: [0..127] logits (64x2), [128..8319] g (64x128)
__global__ __launch_bounds__(256) void head_kernel(const float* __restrict__ sums,
                                                   const float* __restrict__ counts,
                                                   const float* __restrict__ HW,
                                                   const float* __restrict__ HB,
                                                   float* __restrict__ out) {
    __shared__ float sG[NG * HID];
    for (int idx = threadIdx.x; idx < NG * HID; idx += 256) {
        int g = idx >> 7;
        float c = counts[g];
        c = c > 1.f ? c : 1.f;
        float v = sums[idx] / c;
        sG[idx] = v;
        out[NG * NC + idx] = v;
    }
    __syncthreads();
    if (threadIdx.x < NG * NC) {
        int g = threadIdx.x >> 1, cls = threadIdx.x & 1;
        float acc = HB[cls];
        for (int k = 0; k < HID; ++k) acc += sG[g * HID + k] * HW[k * NC + cls];
        out[g * NC + cls] = acc;
    }
}

extern "C" void kernel_launch(void* const* d_in, const int* in_sizes, int n_in,
                              void* d_out, int out_size, void* d_ws, size_t ws_size,
                              hipStream_t stream) {
    const float* X  = (const float*)d_in[0];
    const int*   EI = (const int*)d_in[1];
    const int*   BT = (const int*)d_in[2];
    const float* W1 = (const float*)d_in[3];
    const float* B1 = (const float*)d_in[4];
    const float* W2 = (const float*)d_in[5];
    const float* B2 = (const float*)d_in[6];
    const float* HW = (const float*)d_in[7];
    const float* HB = (const float*)d_in[8];
    float* outp = (float*)d_out;

    const int nNodes = in_sizes[0] / HID;
    const int nEdges = in_sizes[1] / 2;
    const int* src = EI;
    const int* dst = EI + nEdges;

    const size_t NF = (size_t)nNodes * HID;
    float* bufA   = (float*)d_ws;          // activation ping
    float* bufB   = bufA + NF;             // aggregation / pong
    float* sums   = bufB + NF;             // [64,128]
    float* counts = sums + NG * HID;       // [64]

    const int n4        = (int)(NF / 4);
    const int copyGrid  = (n4 + 255) / 256;
    const int scatGrid  = (nEdges * 32 + 255) / 256;
    const int mlpGrid   = (nNodes + 63) / 64;
    const int poolGrid  = (nNodes * 32 + 255) / 256;

    const float* cur = X;
    for (int l = 0; l < NL; ++l) {
        // agg = x  (fuses the "+ x_i" of GIN eps=0)
        copy_vec4<<<copyGrid, 256, 0, stream>>>(cur, bufB, n4);
        // agg[dst] += x[src]
        scatter_edges<<<scatGrid, 256, 0, stream>>>(cur, src, dst, bufB, nEdges);
        // x = relu(relu(agg@W1+b1)@W2+b2)
        gin_mlp<<<mlpGrid, 256, 0, stream>>>(bufB, bufA,
                                             W1 + (size_t)l * HID * HID, B1 + l * HID,
                                             W2 + (size_t)l * HID * HID, B2 + l * HID,
                                             nNodes);
        cur = bufA;
    }

    pool_zero<<<(NG * HID + 255) / 256, 256, 0, stream>>>(sums, counts);
    pool_scatter<<<poolGrid, 256, 0, stream>>>(cur, BT, sums, counts, nNodes);
    head_kernel<<<1, 256, 0, stream>>>(sums, counts, HW, HB, outp);
}
